// cRNN_90718299226158
// MI455X (gfx1250) — matrix operand order repaired and checked by reading, not compiled, not run on hardware
//
#include <hip/hip_runtime.h>

// RNN: B=512, T=2048, I=4, H=100, O=1.
// Latency-bound sequential recurrence -> keep h resident in LDS, weights in
// WMMA B-fragment registers, 8x v_wmma_f32_16x16x32_bf16 per step per wave.

#define BDIM 512
#define TDIM 2048
#define IDIM 4
#define HDIM 100
#define HSTR 136   // padded row stride (ushorts): 272B = 68 dwords -> conflict-free b128 loads

typedef __attribute__((ext_vector_type(16))) __bf16 v16bf;
typedef __attribute__((ext_vector_type(8)))  float  v8f;

union BF16x16 { unsigned u[8]; v16bf v; };
union F32x8   { float f[8]; v8f v; };

__device__ __forceinline__ unsigned short f2bf(float f) {
  unsigned u = __float_as_uint(f);
  u = (u + 0x7FFFu + ((u >> 16) & 1u)) >> 16;   // round-to-nearest-even
  return (unsigned short)u;
}
__device__ __forceinline__ float bf_lo(unsigned u) { return __uint_as_float(u << 16); }
__device__ __forceinline__ float bf_hi(unsigned u) { return __uint_as_float(u & 0xFFFF0000u); }

__device__ __forceinline__ float fast_tanh(float x) {
  x = fminf(9.0f, fmaxf(-9.0f, x));
  float e = __builtin_amdgcn_exp2f(x * 2.8853900817779268f);  // exp(2x)
  return (e - 1.0f) * __builtin_amdgcn_rcpf(e + 1.0f);
}

__device__ __forceinline__ float dot_row_wfc(const unsigned short* hr,
                                             const float* wfc, float bias) {
  float acc = bias;
  #pragma unroll
  for (int n = 0; n < 104; n += 8) {
    const uint4 hv = *reinterpret_cast<const uint4*>(hr + n);
    acc += bf_lo(hv.x) * wfc[n + 0] + bf_hi(hv.x) * wfc[n + 1]
         + bf_lo(hv.y) * wfc[n + 2] + bf_hi(hv.y) * wfc[n + 3]
         + bf_lo(hv.z) * wfc[n + 4] + bf_hi(hv.z) * wfc[n + 5]
         + bf_lo(hv.w) * wfc[n + 6] + bf_hi(hv.w) * wfc[n + 7];
  }
  return acc;
}

__global__ __launch_bounds__(128)
void crnn_fused_kernel(const float* __restrict__ x,
                       const float* __restrict__ W_ih,
                       const float* __restrict__ W_hh,
                       const float* __restrict__ W_fc,
                       const float* __restrict__ b_fc,
                       float* __restrict__ out) {
  // Augmented weight: waug[k][n] = (k<100 ? W_hh[n][k] : k<104 ? W_ih[n][k-100] : 0)
  __shared__ __align__(16) unsigned short waug[128 * 128];      // 32 KB, bf16 [k][n]
  __shared__ __align__(16) unsigned short hbuf[2][16 * HSTR];   // ping-pong [h | x | 0] bf16
  __shared__ __align__(16) float wfc[104];                      // fp32, zero-padded

  const int tid  = threadIdx.x;
  const int wv   = tid >> 5;       // wave id (wave32)
  const int lane = tid & 31;
  const int nloc = lane & 15;
  const int b0   = blockIdx.x * 16;

  // ---------------- init ----------------
  for (int idx = tid; idx < 128 * 128; idx += 128) {
    const int k = idx >> 7, n = idx & 127;
    float v = 0.0f;
    if (n < HDIM) {
      if (k < HDIM)              v = W_hh[n * HDIM + k];
      else if (k < HDIM + IDIM)  v = W_ih[n * IDIM + (k - HDIM)];
    }
    waug[idx] = f2bf(v);
  }
  for (int idx = tid; idx < 2 * 16 * HSTR; idx += 128) (&hbuf[0][0])[idx] = 0;
  for (int idx = tid; idx < 104; idx += 128) wfc[idx] = (idx < HDIM) ? W_fc[idx] : 0.0f;
  __syncthreads();

  // x_0 -> buf0 columns 100..103
  if (wv == 0 && lane < 16) {
    const float4 xv =
        *reinterpret_cast<const float4*>(x + ((size_t)(b0 + lane) * TDIM) * IDIM);
    const unsigned lo = (unsigned)f2bf(xv.x) | ((unsigned)f2bf(xv.y) << 16);
    const unsigned hi = (unsigned)f2bf(xv.z) | ((unsigned)f2bf(xv.w) << 16);
    *reinterpret_cast<uint2*>(&hbuf[0][lane * HSTR + 100]) = make_uint2(lo, hi);
  }
  const float bias = b_fc[0];

  // ---- preload constant B fragments (tiles wv and wv+4; wave3's 2nd tile is zero pad) ----
  const int tn0 = wv, tn1 = wv + 4;
  const int kbB = (lane < 16) ? 0 : 16;   // B, contiguous k order used here (same as A): lanes 0-15 K=0..15, 16-31 K=16..31
  BF16x16 bfr[2][4];
  #pragma unroll
  for (int tt = 0; tt < 2; ++tt) {
    const int n = ((tt == 0) ? tn0 : tn1) * 16 + nloc;
    #pragma unroll
    for (int c = 0; c < 4; ++c) {
      #pragma unroll
      for (int j = 0; j < 8; ++j) {
        const int k0 = c * 32 + kbB + 2 * j;
        const unsigned lo = waug[k0 * 128 + n];
        const unsigned hi = waug[(k0 + 1) * 128 + n];
        bfr[tt][c].u[j] = lo | (hi << 16);
      }
    }
  }
  __syncthreads();

  // ---------------- recurrence ----------------
  const int selA  = (lane < 16) ? 0 : 16; // A, contiguous k order used here (same as B): lanes 0-15 K=0..15, 16-31 K=16..31
  const int mbase = (lane < 16) ? 0 : 8;  // C/D layout: VGPR r -> M = r + mbase

  #pragma unroll 1
  for (int t = 0; t < TDIM; ++t) {
    const int rb = t & 1, wb = rb ^ 1;

    // overlap: stage x_{t+1} into write buffer; prefetch x_{t+2}
    if (wv == 0 && lane < 16 && (t + 1) < TDIM) {
      const float4 xv = *reinterpret_cast<const float4*>(
          x + ((size_t)(b0 + lane) * TDIM + (t + 1)) * IDIM);
      const unsigned lo = (unsigned)f2bf(xv.x) | ((unsigned)f2bf(xv.y) << 16);
      const unsigned hi = (unsigned)f2bf(xv.z) | ((unsigned)f2bf(xv.w) << 16);
      *reinterpret_cast<uint2*>(&hbuf[wb][lane * HSTR + 100]) = make_uint2(lo, hi);
      if ((t + 2) < TDIM)
        __builtin_prefetch(x + ((size_t)(b0 + lane) * TDIM + (t + 2)) * IDIM, 0, 0);
    }

    // A fragments: row m = nloc of [h_{t-1} | x_t | 0], K = 0..127
    const unsigned short* hrow = &hbuf[rb][nloc * HSTR];
    BF16x16 afr[4];
    #pragma unroll
    for (int c = 0; c < 4; ++c) {
      const uint4 a0 = *reinterpret_cast<const uint4*>(hrow + c * 32 + selA);
      const uint4 a1 = *reinterpret_cast<const uint4*>(hrow + c * 32 + selA + 8);
      afr[c].u[0] = a0.x; afr[c].u[1] = a0.y; afr[c].u[2] = a0.z; afr[c].u[3] = a0.w;
      afr[c].u[4] = a1.x; afr[c].u[5] = a1.y; afr[c].u[6] = a1.z; afr[c].u[7] = a1.w;
    }

    // two 16x16 output tiles, K=128 via 4 chained bf16 WMMAs each
    #pragma unroll
    for (int tt = 0; tt < 2; ++tt) {
      F32x8 acc;
      #pragma unroll
      for (int r = 0; r < 8; ++r) acc.f[r] = 0.0f;
      #pragma unroll
      for (int c = 0; c < 4; ++c)
        acc.v = __builtin_amdgcn_wmma_f32_16x16x32_bf16(
            false, afr[c].v, false, bfr[tt][c].v, (short)0, acc.v, false, false);

      const int n = ((tt == 0) ? tn0 : tn1) * 16 + nloc;
      if (n < HDIM) {
        #pragma unroll
        for (int r = 0; r < 8; ++r)
          hbuf[wb][(mbase + r) * HSTR + n] = f2bf(fast_tanh(acc.f[r]));
      }
    }

    // wave 3 overlaps: out[., t-1] = h_{t-1} . W_fc + b  (reads stable rb buffer)
    if (wv == 3 && lane < 16 && t > 0)
      out[(size_t)(b0 + lane) * TDIM + (t - 1)] =
          dot_row_wfc(&hbuf[rb][lane * HSTR], wfc, bias);

    __syncthreads();
  }

  // epilogue: out for t = T-1 (h_{T-1} lives in buffer (T & 1) == 0)
  if (wv == 3 && lane < 16)
    out[(size_t)(b0 + lane) * TDIM + (TDIM - 1)] =
        dot_row_wfc(&hbuf[TDIM & 1][lane * HSTR], wfc, bias);
}

extern "C" void kernel_launch(void* const* d_in, const int* in_sizes, int n_in,
                              void* d_out, int out_size, void* d_ws, size_t ws_size,
                              hipStream_t stream) {
  (void)in_sizes; (void)n_in; (void)out_size; (void)d_ws; (void)ws_size;
  const float* x    = (const float*)d_in[0];
  const float* W_ih = (const float*)d_in[1];
  const float* W_hh = (const float*)d_in[2];
  const float* W_fc = (const float*)d_in[3];
  const float* b_fc = (const float*)d_in[4];
  float* out = (float*)d_out;

  crnn_fused_kernel<<<dim3(BDIM / 16), dim3(128), 0, stream>>>(
      x, W_ih, W_hh, W_fc, b_fc, out);
}
